// Pointnet_Tracking_74577812128447
// MI455X (gfx1250) — matrix operand order repaired and checked by reading, not compiled, not run on hardware
//
#include <hip/hip_runtime.h>
#include <hip/hip_bf16.h>
#include <cstdint>
#include <cstddef>

typedef __attribute__((ext_vector_type(16))) _Float16 v16h;
typedef __attribute__((ext_vector_type(8)))  _Float16 h8v;
typedef __attribute__((ext_vector_type(2)))  _Float16 h2;
typedef __attribute__((ext_vector_type(8)))  float    v8f;
typedef __attribute__((ext_vector_type(4)))  float    f4;
typedef int v4i __attribute__((vector_size(16)));   // matches builtin's int4 pointee

#define DEV __device__ __forceinline__
static constexpr int   BATCH = 16;
static constexpr float BN_EPS = 1e-5f;

#if defined(__has_builtin)
#if __has_builtin(__builtin_amdgcn_global_load_async_to_lds_b128) && \
    __has_builtin(__builtin_amdgcn_s_wait_asynccnt)
#define ASYNC_BUILTIN 1
#endif
#endif

// per-lane async copy of 16B global -> LDS (CDNA5 GLOBAL_LOAD_ASYNC_TO_LDS_B128)
DEV void async_copy_b128(const _Float16* g, _Float16* l) {
#if defined(ASYNC_BUILTIN)
  __builtin_amdgcn_global_load_async_to_lds_b128(
      (__attribute__((address_space(1))) v4i*)g,
      (__attribute__((address_space(3))) v4i*)l, 0, 0);
#else
  unsigned loff = (unsigned)(size_t)(__attribute__((address_space(3))) void*)l;
  asm volatile("global_load_async_to_lds_b128 %0, %1, off"
               :: "v"(loff), "v"(g) : "memory");
#endif
}
DEV void async_wait0() {
#if defined(ASYNC_BUILTIN)
  __builtin_amdgcn_s_wait_asynccnt(0);
#else
  asm volatile("s_wait_asynccnt 0" ::: "memory");
#endif
}

// ----------------------------------------------------------------------------
// Generic WMMA GEMM:  C[M,N] = A[M,K] * B[K,N] (+bias)
//   A: f32 [M][K] weights (converted to f16 at staging)
//   B: mode 4: f16 activations [n][K] (NHWC)   -> async-to-LDS fast path
//      mode 2: f32 implicit im2col (3-D conv)
//      mode 3: f32 implicit im2col (2-D conv, channel = c*Dsplit+d decode)
//   C: f32 ; outT=1 -> C[n*ldC+m] (NHWC), outT=0 -> C[m*ldC+n]
//   Double-buffered LDS pipeline: stage tile t+1 (async DMA for mode 4)
//   while WMMAs consume tile t; one s_wait_asynccnt + one barrier per K-step.
// ----------------------------------------------------------------------------
struct GemmP {
  const float*    A;
  const float*    Bsrc;   // f32 source for conv modes
  const _Float16* B16;    // f16 source for mode 4
  const float*    bias;
  float*          C;
  int M, N, K;
  int mode;
  int outT;
  long long ldB, ldC;
  // conv params
  int Bn;
  int Din, Hin, Win;
  int Dout, Hout, Wout;
  int sd, pd;
  int Dsplit;
};

DEV float fetch_conv(const GemmP& p, int k, int n) {
  if (k >= p.K || n >= p.N) return 0.f;
  if (p.mode == 2) {
    int ci = k / 27, r = k % 27;
    int kd = r / 9; r %= 9;
    int kh = r / 3, kw = r % 3;
    int sp = p.Dout * p.Hout * p.Wout;
    int b  = n / sp, r2 = n % sp;
    int od = r2 / (p.Hout * p.Wout); r2 %= (p.Hout * p.Wout);
    int oh = r2 / p.Wout, ow = r2 % p.Wout;
    int id = od * p.sd - p.pd + kd;
    int ih = oh - 1 + kh;
    int iw = ow - 1 + kw;
    if (id < 0 || id >= p.Din || ih < 0 || ih >= p.Hin || iw < 0 || iw >= p.Win) return 0.f;
    long long a = (((long long)ci * p.Bn + b) * p.Din + id) * (long long)(p.Hin * p.Win)
                + (long long)ih * p.Win + iw;
    return p.Bsrc[a];
  }
  int ci = k / 9, r = k % 9;
  int kh = r / 3, kw = r % 3;
  int sp = p.Hout * p.Wout;
  int b  = n / sp, r2 = n % sp;
  int oh = r2 / p.Wout, ow = r2 % p.Wout;
  int ih = oh - 1 + kh, iw = ow - 1 + kw;
  if (ih < 0 || ih >= p.Hin || iw < 0 || iw >= p.Win) return 0.f;
  int c = ci / p.Dsplit, d = ci % p.Dsplit;
  long long a = (((long long)c * p.Bn + b) * p.Dsplit + d) * (long long)(p.Hin * p.Win)
              + (long long)ih * p.Win + iw;
  return p.Bsrc[a];
}

DEV h8v pack8(f4 a0, f4 a1) {
  h8v h;
  h[0] = (_Float16)a0.x; h[1] = (_Float16)a0.y;
  h[2] = (_Float16)a0.z; h[3] = (_Float16)a0.w;
  h[4] = (_Float16)a1.x; h[5] = (_Float16)a1.y;
  h[6] = (_Float16)a1.z; h[7] = (_Float16)a1.w;
  return h;
}

__global__ __launch_bounds__(256)
void wmma_gemm_kernel(GemmP p) {
  __shared__ __align__(16) _Float16 As[2][64][32];   // [buf][m][k]
  __shared__ __align__(16) _Float16 Bs[2][128][32];  // [buf][n][k]
  const int tid  = threadIdx.x;
  const int lane = tid & 31;
  const int wave = tid >> 5;
  const int wm   = wave & 3;
  const int wn   = wave >> 2;
  const int nBase = blockIdx.x * 128;
  const int mBase = blockIdx.y * 64;

  v8f acc[4];
#pragma unroll
  for (int s = 0; s < 4; ++s)
#pragma unroll
    for (int r = 0; r < 8; ++r) acc[s][r] = 0.f;

  const bool a_fast_m = (mBase + 64 <= p.M) && ((p.K & 7) == 0);

  auto stage = [&](int k0, int buf) {
    const bool kfull = (k0 + 32 <= p.K);
    // ---- A tile (64x32 f32->f16) ----
    if (a_fast_m && kfull) {
      int m = tid >> 2, kc = (tid & 3) * 8;
      const float* ap = p.A + (long long)(mBase + m) * p.K + k0 + kc;
      f4 a0 = *(const f4*)ap;
      f4 a1 = *(const f4*)(ap + 4);
      *(h8v*)&As[buf][m][kc] = pack8(a0, a1);
    } else {
#pragma unroll
      for (int j = 0; j < 8; ++j) {
        int li = j * 256 + tid;
        int m = li >> 5, k = li & 31;
        int gm = mBase + m, gk = k0 + k;
        float v = (gm < p.M && gk < p.K) ? p.A[(long long)gm * p.K + gk] : 0.f;
        As[buf][m][k] = (_Float16)v;
      }
    }
    // ---- B tile (128 rows x 32 halves) ----
    if (p.mode == 4) {
      bool bfast = kfull && (nBase + 128 <= p.N) && ((p.ldB & 7) == 0);
      if (bfast) {
#pragma unroll
        for (int is = 0; is < 2; ++is) {
          int li = is * 256 + tid;
          int n = li >> 2, kc = (li & 3) * 8;
          const _Float16* gp = p.B16 + (long long)(nBase + n) * p.ldB + (k0 + kc);
          async_copy_b128(gp, &Bs[buf][n][kc]);
        }
        if (k0 + 32 < p.K)
          __builtin_prefetch(p.B16 + (long long)(nBase + lane) * p.ldB + k0 + 32, 0, 1);
      } else {
#pragma unroll
        for (int j = 0; j < 16; ++j) {
          int li = j * 256 + tid;
          int n = li >> 5, k = li & 31;
          int gk = k0 + k, gn = nBase + n;
          Bs[buf][n][k] = (gk < p.K && gn < p.N)
                              ? p.B16[(long long)gn * p.ldB + gk] : (_Float16)0.f;
        }
      }
    } else {
#pragma unroll
      for (int j = 0; j < 16; ++j) {
        int li = j * 256 + tid;
        int n = li >> 5, k = li & 31;
        Bs[buf][n][k] = (_Float16)fetch_conv(p, k0 + k, nBase + n);
      }
    }
  };

  const int nT = (p.K + 31) >> 5;
  stage(0, 0);
  async_wait0();
  __syncthreads();

  for (int t = 0; t < nT; ++t) {
    const int buf = t & 1;
    if (t + 1 < nT) stage((t + 1) * 32, buf ^ 1);  // overlap DMA with WMMA

    // ---- fragments + WMMA on tile t ----
    v16h af;
    {
      int am = wm * 16 + (lane & 15);
      int hi = (lane >> 4) << 3;
#pragma unroll
      for (int q = 0; q < 8; ++q) {
        int kb = ((q < 4) ? (2 * q) : (16 + 2 * (q - 4))) + hi;
        h2 pr = *reinterpret_cast<const h2*>(&As[buf][am][kb]);
        af[2 * q]     = pr.x;
        af[2 * q + 1] = pr.y;
      }
    }
    int bn0 = wn * 64 + (lane & 15);
    int kbb = (lane >> 4) << 3;
#pragma unroll
    for (int s = 0; s < 4; ++s) {
      v16h bf;
#pragma unroll
      for (int q = 0; q < 8; ++q) {
        h2 pr = *reinterpret_cast<const h2*>(&Bs[buf][bn0 + s * 16][kbb + ((q < 4) ? (2 * q) : (16 + 2 * (q - 4)))]);
        bf[2 * q]     = pr.x;
        bf[2 * q + 1] = pr.y;
      }
      acc[s] = __builtin_amdgcn_wmma_f32_16x16x32_f16(
          false, af, false, bf, (short)0, acc[s], false, false);
    }

    async_wait0();       // this wave's copies for tile t+1 landed
    __syncthreads();     // all waves' staging of t+1 visible; frag reads of t done
  }

#pragma unroll
  for (int s = 0; s < 4; ++s) {
#pragma unroll
    for (int r = 0; r < 8; ++r) {
      int m = mBase + wm * 16 + r + ((lane >> 4) << 3);
      int n = nBase + wn * 64 + s * 16 + (lane & 15);
      if (m < p.M && n < p.N) {
        float v = acc[s][r];
        if (p.bias) v += p.bias[m];
        if (p.outT) p.C[(long long)n * p.ldC + m] = v;
        else        p.C[(long long)m * p.ldC + n] = v;
      }
    }
  }
}

// ----------------------------------------------------------------------------
// BatchNorm, channel-row layout (conv outputs [m][N], f32 in-place)
// ----------------------------------------------------------------------------
__global__ void bn_stats_rows_kernel(const float* y, long long R, float* mean,
                                     float* rstd) {
  __shared__ float ssum[256], ssq[256];
  int c = blockIdx.x;
  const float* row = y + (long long)c * R;
  float s = 0.f, q = 0.f;
  for (long long i = threadIdx.x; i < R; i += 256) {
    float v = row[i];
    s += v; q += v * v;
  }
  ssum[threadIdx.x] = s; ssq[threadIdx.x] = q;
  __syncthreads();
  for (int off = 128; off > 0; off >>= 1) {
    if (threadIdx.x < off) {
      ssum[threadIdx.x] += ssum[threadIdx.x + off];
      ssq[threadIdx.x]  += ssq[threadIdx.x + off];
    }
    __syncthreads();
  }
  if (threadIdx.x == 0) {
    float m = ssum[0] / (float)R;
    float v = ssq[0] / (float)R - m * m;
    mean[c] = m;
    rstd[c] = rsqrtf(fmaxf(v, 0.f) + BN_EPS);
  }
}

__global__ void bn_apply_rows_kernel(float* y, long long R, int C,
                                     const float* mean, const float* rstd,
                                     const float* g, const float* b) {
  long long i = (long long)blockIdx.x * blockDim.x + threadIdx.x;
  long long total = (long long)C * R;
  if (i >= total) return;
  int c = (int)(i / R);
  float v = (y[i] - mean[c]) * rstd[c] * g[c] + b[c];
  y[i] = fmaxf(v, 0.f);
}

// ----------------------------------------------------------------------------
// BatchNorm, NHWC layout (MLP GEMM outputs [n][C] f32 -> f16 mirror)
// C must divide 256 (C = 64 or 128 here)
// ----------------------------------------------------------------------------
__global__ void bn_nhwc_partial_kernel(const float* y, long long Nrows, int C,
                                       float* psum, float* psq) {
  __shared__ float ls[256], lq[256];
  int t = threadIdx.x;
  int rpb = 256 / C;
  int c  = t % C;
  int ro = t / C;
  float s = 0.f, q = 0.f;
  for (long long r = (long long)blockIdx.x * rpb + ro; r < Nrows;
       r += (long long)gridDim.x * rpb) {
    float v = y[r * C + c];
    s += v; q += v * v;
  }
  ls[t] = s; lq[t] = q;
  __syncthreads();
  if (ro == 0) {
    for (int o = 1; o < rpb; ++o) { s += ls[o * C + c]; q += lq[o * C + c]; }
    atomicAdd(&psum[c], s);
    atomicAdd(&psq[c], q);
  }
}

__global__ void bn_finalize_kernel(const float* psum, const float* psq,
                                   long long cnt, int C, float* mean, float* rstd) {
  int c = blockIdx.x * blockDim.x + threadIdx.x;
  if (c >= C) return;
  float m = psum[c] / (float)cnt;
  float v = psq[c] / (float)cnt - m * m;
  mean[c] = m;
  rstd[c] = rsqrtf(fmaxf(v, 0.f) + BN_EPS);
}

__global__ void bn_apply_nhwc_kernel(const float* y, long long Nrows, int C,
                                     const float* mean, const float* rstd,
                                     const float* g, const float* b,
                                     _Float16* out16) {
  long long i = (long long)blockIdx.x * blockDim.x + threadIdx.x;
  long long total = Nrows * C;
  if (i >= total) return;
  int c = (int)(i % C);
  float v = (y[i] - mean[c]) * rstd[c] * g[c] + b[c];
  out16[i] = (_Float16)fmaxf(v, 0.f);
}

// stacks: y laid out [n*F + o], n = b*P + p; BN per point p over (b,o);
// ReLU applied BEFORE normalization (torch order in reference).
__global__ void bn_stats_pts_kernel(const float* y, int P, int F,
                                    float* mean, float* rstd) {
  __shared__ float ssum[256], ssq[256];
  int p = blockIdx.x;
  int total = BATCH * F;
  float s = 0.f, q = 0.f;
  for (int e = threadIdx.x; e < total; e += 256) {
    int b = e / F, o = e % F;
    float v = fmaxf(y[((long long)(b * P + p)) * F + o], 0.f);
    s += v; q += v * v;
  }
  ssum[threadIdx.x] = s; ssq[threadIdx.x] = q;
  __syncthreads();
  for (int off = 128; off > 0; off >>= 1) {
    if (threadIdx.x < off) {
      ssum[threadIdx.x] += ssum[threadIdx.x + off];
      ssq[threadIdx.x]  += ssq[threadIdx.x + off];
    }
    __syncthreads();
  }
  if (threadIdx.x == 0) {
    float m = ssum[0] / (float)total;
    float v = ssq[0] / (float)total - m * m;
    mean[p] = m;
    rstd[p] = rsqrtf(fmaxf(v, 0.f) + BN_EPS);
  }
}

__global__ void bn_apply_pts_kernel(const float* y, int P, int F,
                                    const float* mean, const float* rstd,
                                    const float* g, const float* bb,
                                    _Float16* out16) {
  long long i = (long long)blockIdx.x * blockDim.x + threadIdx.x;
  long long total = (long long)BATCH * P * F;
  if (i >= total) return;
  int n = (int)(i / F);
  int p = n % P;
  float v = fmaxf(y[i], 0.f);
  out16[i] = (_Float16)((v - mean[p]) * rstd[p] * g[p] + bb[p]);
}

// ----------------------------------------------------------------------------
// Farthest point sampling: one block per batch, dist & points in LDS
// ----------------------------------------------------------------------------
__global__ __launch_bounds__(256)
void fps_kernel(const float* xyz, int N, int S, int* idx) {
  __shared__ float px[1024], py[1024], pz[1024], dist[1024];
  __shared__ float rv[256];
  __shared__ int   ri[256];
  __shared__ int   s_last;
  int b = blockIdx.x, t = threadIdx.x;
  const float* base = xyz + (long long)b * N * 3;
  for (int i = t; i < N; i += 256) {
    px[i] = base[i * 3 + 0];
    py[i] = base[i * 3 + 1];
    pz[i] = base[i * 3 + 2];
    dist[i] = 1e10f;
  }
  if (t == 0) { s_last = 0; idx[b * S] = 0; }
  __syncthreads();
  for (int it = 1; it < S; ++it) {
    int last = s_last;
    float lx = px[last], ly = py[last], lz = pz[last];
    float best = -1.f; int bi = 0;
    for (int i = t; i < N; i += 256) {
      float dx = px[i] - lx, dy = py[i] - ly, dz = pz[i] - lz;
      float d = dx * dx + dy * dy + dz * dz;
      float dm = fminf(dist[i], d);
      dist[i] = dm;
      if (dm > best) { best = dm; bi = i; }
    }
    rv[t] = best; ri[t] = bi;
    __syncthreads();
    for (int off = 128; off > 0; off >>= 1) {
      if (t < off) {
        float v2 = rv[t + off]; int i2 = ri[t + off];
        if (v2 > rv[t] || (v2 == rv[t] && i2 < ri[t])) { rv[t] = v2; ri[t] = i2; }
      }
      __syncthreads();
    }
    if (t == 0) { s_last = ri[0]; idx[b * S + it] = ri[0]; }
    __syncthreads();
  }
}

__global__ void gather_xyz_kernel(const float* xyz, const int* idx, int N, int S,
                                  float* out) {
  int i = blockIdx.x * blockDim.x + threadIdx.x;
  int total = BATCH * S * 3;
  if (i >= total) return;
  int c = i % 3, bs = i / 3;
  int b = bs / S;
  int j = idx[bs];
  out[i] = xyz[((long long)b * N + j) * 3 + c];
}

__global__ void ball_query_kernel(const float* xyz, const float* cent, int N, int S,
                                  int ns, float r2, int* out) {
  int i = blockIdx.x * blockDim.x + threadIdx.x;
  if (i >= BATCH * S) return;
  int b = i / S;
  float cx = cent[i * 3], cy = cent[i * 3 + 1], cz = cent[i * 3 + 2];
  const float* base = xyz + (long long)b * N * 3;
  int* o = out + (long long)i * ns;
  int cnt = 0, first = 0; bool found = false;
  for (int j = 0; j < N && cnt < ns; ++j) {
    float dx = base[j * 3] - cx, dy = base[j * 3 + 1] - cy, dz = base[j * 3 + 2] - cz;
    if (dx * dx + dy * dy + dz * dz < r2) {
      if (!found) { found = true; first = j; }
      o[cnt++] = j;
    }
  }
  for (; cnt < ns; ++cnt) o[cnt] = found ? first : 0;
}

// grouping writes the f16 NHWC GEMM operand directly: g16[r][C]
__global__ void group_xyz_kernel(const float* xyz, const float* cent, const int* nidx,
                                 int N, int S, int ns, _Float16* g16) {
  long long R = (long long)BATCH * S * ns;
  long long i = (long long)blockIdx.x * blockDim.x + threadIdx.x;
  if (i >= R) return;
  long long bs = i / ns;
  int b = (int)(bs / S);
  int j = nidx[i];
  const float* pj = xyz + ((long long)b * N + j) * 3;
  const float* pc = cent + bs * 3;
  g16[i * 3 + 0] = (_Float16)(pj[0] - pc[0]);
  g16[i * 3 + 1] = (_Float16)(pj[1] - pc[1]);
  g16[i * 3 + 2] = (_Float16)(pj[2] - pc[2]);
}

__global__ void group_feat_kernel(const float* f, const int* nidx, int Nprev, int S,
                                  int ns, int C, _Float16* g16) {
  long long R = (long long)BATCH * S * ns;
  long long i = (long long)blockIdx.x * blockDim.x + threadIdx.x;
  long long total = R * C;
  if (i >= total) return;
  long long rr = i / C;
  int c = (int)(i % C);
  long long bs = rr / ns;
  int b = (int)(bs / S);
  int j = nidx[rr];
  g16[i] = (_Float16)f[((long long)c * BATCH + b) * Nprev + j];
}

// max over ns on the post-BN f16 mirror [r][C]; features out f32 [c][B*S]
__global__ void maxpool_ns_kernel(const _Float16* y16, int S, int ns, int C,
                                  float* f) {
  int i = blockIdx.x * blockDim.x + threadIdx.x;
  int total = C * BATCH * S;
  if (i >= total) return;
  int bs = i % (BATCH * S);
  int c  = i / (BATCH * S);
  const _Float16* row = y16 + (long long)bs * ns * C + c;
  float m = (float)row[0];
  for (int n = 1; n < ns; ++n) m = fmaxf(m, (float)row[(long long)n * C]);
  f[i] = m;
}

__global__ void group_max_kernel(const float* xyz, const float* cent, const float* f,
                                 const int* nidx, int N, int P, int ns, float* out) {
  int i = blockIdx.x * blockDim.x + threadIdx.x;
  int NP = BATCH * P;
  int total = 131 * NP;
  if (i >= total) return;
  int c  = i / NP;
  int bp = i % NP;
  int b  = bp / P;
  const int* nb = nidx + (long long)bp * ns;
  float m = -3.4e38f;
  for (int n = 0; n < ns; ++n) {
    int j = nb[n];
    float v;
    if (c < 3) v = xyz[((long long)b * N + j) * 3 + c] - cent[(long long)bp * 3 + c];
    else       v = f[((long long)(c - 3) * BATCH + b) * N + j];
    m = fmaxf(m, v);
  }
  out[i] = m;
}

// ----------------------------------------------------------------------------
// misc elementwise / assembly kernels
// ----------------------------------------------------------------------------
__global__ void zero_kernel(float* p, long long n) {
  long long i = (long long)blockIdx.x * blockDim.x + threadIdx.x;
  if (i < n) p[i] = 0.f;
}

__global__ void sigmoid_kernel(const float* in, float* out, int n) {
  int i = blockIdx.x * blockDim.x + threadIdx.x;
  if (i < n) out[i] = 1.f / (1.f + expf(-in[i]));
}

__global__ void rows_to_stack_kernel(const float* src, int NN, int K, _Float16* x) {
  long long i = (long long)blockIdx.x * blockDim.x + threadIdx.x;
  long long total = (long long)NN * K;
  if (i >= total) return;
  int n = (int)(i / K), k = (int)(i % K);
  x[i] = (_Float16)src[(long long)k * NN + n];
}

__global__ void assemble_fx_kernel(const float* sxyz, const float* f3, int P,
                                   _Float16* x) {
  int NN = BATCH * P;
  long long i = (long long)blockIdx.x * blockDim.x + threadIdx.x;
  long long total = (long long)NN * 131;
  if (i >= total) return;
  int n = (int)(i / 131), k = (int)(i % 131);
  float v = (k < 3) ? sxyz[n * 3 + k] : f3[(long long)(k - 3) * NN + n];
  x[i] = (_Float16)v;
}

__global__ void temp_sel_kernel(const float* off, const float* sxyz, int P, float* ts) {
  int NN = BATCH * P;
  int i = blockIdx.x * blockDim.x + threadIdx.x;
  if (i >= NN * 3) return;
  int n = i / 3, c = i % 3;
  ts[i] = sxyz[i] - off[(long long)n * 131 + c];
}

__global__ void fusion_update_kernel(const float* off, const float* f3, int P,
                                     float* fusion) {
  int NN = BATCH * P;
  long long i = (long long)blockIdx.x * blockDim.x + threadIdx.x;
  if (i >= (long long)128 * NN) return;
  int n = (int)(i % NN);
  fusion[i] = f3[i] + off[(long long)n * 131 + 3 + (int)(i / NN)];
}

__global__ void assemble_prop_kernel(const float* score, const float* tp,
                                     const float* sp, const float* fusion, int P,
                                     _Float16* x) {
  int NN = BATCH * P;
  long long i = (long long)blockIdx.x * blockDim.x + threadIdx.x;
  long long total = (long long)NN * 391;
  if (i >= total) return;
  int n = (int)(i / 391), k = (int)(i % 391);
  float v;
  if (k == 0)        v = score[n];
  else if (k < 132)  v = tp[(long long)(k - 1) * NN + n];
  else if (k < 263)  v = sp[(long long)(k - 132) * NN + n];
  else               v = fusion[(long long)(k - 263) * NN + n];
  x[i] = (_Float16)v;
}

// voxelize: scatter-mean of (prop ++ xyz) into (C=131, B, Z=18, Y=24, X=38)
__global__ void vox_scatter_kernel(const float* prop, const float* sxyz, int P,
                                   float* vox, float* cnt) {
  int i = blockIdx.x * blockDim.x + threadIdx.x;
  int total = BATCH * P * 132;
  if (i >= total) return;
  int c  = i % 132;
  int bp = i / 132;
  int b  = bp / P;
  float X = sxyz[bp * 3 + 0], Y = sxyz[bp * 3 + 1], Z = sxyz[bp * 3 + 2];
  int vx = (int)floorf((X + 5.6f) / 0.3f);
  int vy = (int)floorf((Y + 3.6f) / 0.3f);
  int vz = (int)floorf((Z + 2.4f) / 0.3f);
  vx = min(max(vx, 0), 37); vy = min(max(vy, 0), 23); vz = min(max(vz, 0), 17);
  long long pos = ((long long)vz * 24 + vy) * 38 + vx;
  if (c == 131) { atomicAdd(&cnt[(long long)b * 16416 + pos], 1.f); return; }
  float v = (c < 128) ? prop[(long long)bp * 128 + c] : sxyz[bp * 3 + (c - 128)];
  atomicAdd(&vox[((long long)c * BATCH + b) * 16416 + pos], v);
}

__global__ void vox_norm_kernel(float* vox, const float* cnt) {
  long long i = (long long)blockIdx.x * blockDim.x + threadIdx.x;
  long long total = (long long)131 * BATCH * 16416;
  if (i >= total) return;
  long long rem = i % ((long long)BATCH * 16416);
  vox[i] = vox[i] / fmaxf(cnt[rem], 1.f);
}

__global__ void rpn_heads_kernel(const float* h, const float* w_hm, const float* b_hm,
                                 const float* w_loc, const float* b_loc,
                                 const float* w_z, const float* b_z, float* out) {
  int i = blockIdx.x * blockDim.x + threadIdx.x;
  int total = BATCH * 5 * 912;
  if (i >= total) return;
  int s = i % 912, rest = i / 912;
  int ch = rest % 5, b = rest / 5;
  int oh = s / 38, ow = s % 38;
  const float* w; float bias; int co;
  if (ch == 0)      { w = w_hm;  bias = b_hm[0];       co = 0; }
  else if (ch <= 3) { w = w_loc; bias = b_loc[ch - 1]; co = ch - 1; }
  else              { w = w_z;   bias = b_z[0];        co = 0; }
  float acc = bias;
  for (int c = 0; c < 128; ++c) {
    const float* hb = h + ((long long)c * BATCH + b) * 912;
    const float* wc = w + ((long long)co * 128 + c) * 9;
    for (int kh = 0; kh < 3; ++kh) {
      int ih = oh - 1 + kh; if (ih < 0 || ih >= 24) continue;
      for (int kw = 0; kw < 3; ++kw) {
        int iw = ow - 1 + kw; if (iw < 0 || iw >= 38) continue;
        acc += hb[ih * 38 + iw] * wc[kh * 3 + kw];
      }
    }
  }
  if (ch == 0) acc = 1.f / (1.f + expf(-acc));
  out[i] = acc;
}

// ----------------------------------------------------------------------------
// host-side orchestration
// ----------------------------------------------------------------------------
static inline int cdiv(long long a, long long b) { return (int)((a + b - 1) / b); }

static void launch_gemm(hipStream_t st, const GemmP& p) {
  dim3 grid(cdiv(p.N, 128), cdiv(p.M, 64));
  wmma_gemm_kernel<<<grid, 256, 0, st>>>(p);
}

static void gemm_f16(hipStream_t st, const float* A, const _Float16* B16,
                     const float* bias, float* C, int M, int N, int K,
                     long long ldB, long long ldC) {
  GemmP p{}; p.A = A; p.B16 = B16; p.bias = bias; p.C = C;
  p.M = M; p.N = N; p.K = K; p.mode = 4; p.outT = 1; p.ldB = ldB; p.ldC = ldC;
  launch_gemm(st, p);
}

static void bn_rows(hipStream_t st, float* y, int C, long long R, const float* g,
                    const float* b, float* mean, float* rstd) {
  bn_stats_rows_kernel<<<C, 256, 0, st>>>(y, R, mean, rstd);
  bn_apply_rows_kernel<<<cdiv((long long)C * R, 256), 256, 0, st>>>(
      y, R, C, mean, rstd, g, b);
}

static void bn_nhwc(hipStream_t st, const float* yF, long long Nrows, int C,
                    const float* g, const float* b, float* psum, float* psq,
                    float* mean, float* rstd, _Float16* out16) {
  zero_kernel<<<1, 256, 0, st>>>(psum, C);
  zero_kernel<<<1, 256, 0, st>>>(psq, C);
  int rpb = 256 / C;
  long long nb = (Nrows + rpb - 1) / rpb;
  int blocks = (int)(nb < 1024 ? nb : 1024);
  bn_nhwc_partial_kernel<<<blocks, 256, 0, st>>>(yF, Nrows, C, psum, psq);
  bn_finalize_kernel<<<1, 128, 0, st>>>(psum, psq, Nrows, C, mean, rstd);
  bn_apply_nhwc_kernel<<<cdiv(Nrows * C, 256), 256, 0, st>>>(
      yF, Nrows, C, mean, rstd, g, b, out16);
}

struct SaParams {
  const float* w[3]; const float* bg[3]; const float* bb[3];
  int cin[3]; int cout[3];
};

static void run_sa(hipStream_t st, const float* xyz, const float* feat, int N, int S,
                   float radius, bool xyz_only, const SaParams& sp, float* new_xyz,
                   float* feat_out, int* idx_buf, int* nidx_buf, float* mean,
                   float* rstd, float* psum, float* psq, float* yF,
                   _Float16* s16A, _Float16* s16B) {
  const int ns = 32;
  fps_kernel<<<BATCH, 256, 0, st>>>(xyz, N, S, idx_buf);
  gather_xyz_kernel<<<cdiv((long long)BATCH * S * 3, 256), 256, 0, st>>>(
      xyz, idx_buf, N, S, new_xyz);
  ball_query_kernel<<<cdiv((long long)BATCH * S, 128), 128, 0, st>>>(
      xyz, new_xyz, N, S, ns, radius * radius, nidx_buf);
  long long R = (long long)BATCH * S * ns;
  if (xyz_only) {
    group_xyz_kernel<<<cdiv(R, 256), 256, 0, st>>>(xyz, new_xyz, nidx_buf, N, S, ns, s16A);
  } else {
    group_feat_kernel<<<cdiv(R * sp.cin[0], 256), 256, 0, st>>>(
        feat, nidx_buf, N, S, ns, sp.cin[0], s16A);
  }
  const _Float16* cur = s16A;
  _Float16* mir[2] = {s16B, s16A};
  for (int l = 0; l < 3; ++l) {
    gemm_f16(st, sp.w[l], cur, nullptr, yF, sp.cout[l], (int)R, sp.cin[l],
             sp.cin[l], sp.cout[l]);
    bn_nhwc(st, yF, R, sp.cout[l], sp.bg[l], sp.bb[l], psum, psq, mean, rstd,
            mir[l & 1]);
    cur = mir[l & 1];
  }
  maxpool_ns_kernel<<<cdiv((long long)sp.cout[2] * BATCH * S, 256), 256, 0, st>>>(
      cur, S, ns, sp.cout[2], feat_out);
}

static void run_stack(hipStream_t st, const _Float16* x16, int K0, int NN, int P,
                      const float* const* lw, const float* const* lb,
                      const int* louts, int nlin, const float* const* bg,
                      const float* const* bb, _Float16* m16a, _Float16* m16b,
                      float* yF, float* out, float* mean, float* rstd) {
  const _Float16* cur = x16; int K = K0;
  _Float16* mir[2] = {m16a, m16b};
  for (int i = 0; i < nlin - 1; ++i) {
    gemm_f16(st, lw[i], cur, lb[i], yF, louts[i], NN, K, K, louts[i]);
    bn_stats_pts_kernel<<<P, 256, 0, st>>>(yF, P, louts[i], mean, rstd);
    bn_apply_pts_kernel<<<cdiv((long long)NN * louts[i], 256), 256, 0, st>>>(
        yF, P, louts[i], mean, rstd, bg[i], bb[i], mir[i & 1]);
    cur = mir[i & 1]; K = louts[i];
  }
  gemm_f16(st, lw[nlin - 1], cur, lb[nlin - 1], out, louts[nlin - 1], NN, K, K,
           louts[nlin - 1]);
}

extern "C" void kernel_launch(void* const* d_in, const int* in_sizes, int n_in,
                              void* d_out, int out_size, void* d_ws, size_t ws_size,
                              hipStream_t stream) {
  (void)in_sizes; (void)n_in; (void)out_size; (void)ws_size;
  hipStream_t st = stream;
  const float* tmpl = (const float*)d_in[0];
  const float* srch = (const float*)d_in[1];

  int ii = 2;
  auto nx = [&]() -> const float* { return (const float*)d_in[ii++]; };
  const float *cml_bn_b[3], *cml_bn_g[3], *cml_cb[3], *cml_cw[3];
  for (int l = 0; l < 3; ++l) { cml_bn_b[l]=nx(); cml_bn_g[l]=nx(); cml_cb[l]=nx(); cml_cw[l]=nx(); }
  const float *cla_bb[3], *cla_bg[3];
  for (int l = 0; l < 3; ++l) { cla_bb[l]=nx(); cla_bg[l]=nx(); }
  const float *cla_lb[4], *cla_lw[4];
  for (int l = 0; l < 4; ++l) { cla_lb[l]=nx(); cla_lw[l]=nx(); }
  const float *prop_bb[5], *prop_bg[5];
  for (int l = 0; l < 5; ++l) { prop_bb[l]=nx(); prop_bg[l]=nx(); }
  const float *prop_lb[6], *prop_lw[6];
  for (int l = 0; l < 6; ++l) { prop_lb[l]=nx(); prop_lw[l]=nx(); }
  const float *hm_b=nx(), *hm_w=nx(), *loc_b=nx(), *loc_w=nx();
  const float *sh_bb=nx(), *sh_bg=nx(), *sh_cb=nx(), *sh_cw=nx();
  const float *z_b=nx(), *z_w=nx();
  const float *sa_bb[3][3], *sa_bg[3][3], *sa_lb[3][3], *sa_lw[3][3];
  for (int m = 0; m < 3; ++m)
    for (int l = 0; l < 3; ++l) { sa_bb[m][l]=nx(); sa_bg[m][l]=nx(); sa_lb[m][l]=nx(); sa_lw[m][l]=nx(); }
  const float *vote_bb[4], *vote_bg[4];
  for (int l = 0; l < 4; ++l) { vote_bb[l]=nx(); vote_bg[l]=nx(); }
  const float *vote_lb[5], *vote_lw[5];
  for (int l = 0; l < 5; ++l) { vote_lb[l]=nx(); vote_lw[l]=nx(); }

  char* ws = (char*)d_ws;
  size_t off = 0;
  auto alloc_f = [&](size_t n) -> float* {
    float* p = (float*)(ws + off); off += ((n * 4 + 255) / 256) * 256; return p;
  };
  auto alloc_h = [&](size_t n) -> _Float16* {
    _Float16* p = (_Float16*)(ws + off); off += ((n * 2 + 255) / 256) * 256; return p;
  };
  auto alloc_i = [&](size_t n) -> int* {
    int* p = (int*)(ws + off); off += ((n * 4 + 255) / 256) * 256; return p;
  };

  const int NN = BATCH * 128;
  float* t_x1 = alloc_f(BATCH * 256 * 3);
  float* t_x2 = alloc_f(BATCH * 128 * 3);
  float* t_x3 = alloc_f(BATCH * 64 * 3);
  float* t_f1 = alloc_f((size_t)128 * BATCH * 256);
  float* t_f2 = alloc_f((size_t)128 * BATCH * 128);
  float* t_f3 = alloc_f((size_t)128 * BATCH * 64);
  float* s_x1 = alloc_f(BATCH * 512 * 3);
  float* s_x2 = alloc_f(BATCH * 256 * 3);
  float* s_x3 = alloc_f(BATCH * 128 * 3);
  float* s_f1 = alloc_f((size_t)128 * BATCH * 512);
  float* s_f2 = alloc_f((size_t)128 * BATCH * 256);
  float* s_f3 = alloc_f((size_t)128 * BATCH * 128);
  int*   idx_buf  = alloc_i(BATCH * 512);
  int*   nidx_buf = alloc_i((size_t)BATCH * 512 * 32);
  float* mean = alloc_f(256);
  float* rstd = alloc_f(256);
  float* psum = alloc_f(256);
  float* psq  = alloc_f(256);
  float* stackF = alloc_f((size_t)NN * 128);
  _Float16* stack16_a = alloc_h((size_t)NN * 391);
  _Float16* stack16_b = alloc_h((size_t)NN * 128);
  _Float16* stack16_c = alloc_h((size_t)NN * 128);
  float* cla_y   = alloc_f(NN);
  float* score   = alloc_f(NN);
  float* off_y   = alloc_f((size_t)NN * 131);
  float* ts_buf  = alloc_f((size_t)NN * 3);
  float* fusion  = alloc_f((size_t)128 * NN);
  float* tp_buf  = alloc_f((size_t)131 * NN);
  float* sp_buf  = alloc_f((size_t)131 * NN);
  float* prop_y  = alloc_f((size_t)NN * 128);
  float* cnt_buf = alloc_f((size_t)BATCH * 16416);
  const size_t SLAB = 140ull << 20;
  float* slabA = (float*)(ws + off); off += SLAB;
  float* slabB = (float*)(ws + off); off += SLAB;
  const size_t SLAB16 = 72ull << 20;
  _Float16* s16A = (_Float16*)(ws + off); off += SLAB16;
  _Float16* s16B = (_Float16*)(ws + off); off += SLAB16;

  auto mk_sa = [&](int m, int c0, int c1, int c2, int c3) {
    SaParams s;
    s.cin[0] = c0; s.cout[0] = c1; s.cin[1] = c1; s.cout[1] = c2;
    s.cin[2] = c2; s.cout[2] = c3;
    for (int l = 0; l < 3; ++l) { s.w[l]=sa_lw[m][l]; s.bg[l]=sa_bg[m][l]; s.bb[l]=sa_bb[m][l]; }
    return s;
  };
  SaParams P1 = mk_sa(0, 3, 64, 64, 128);
  SaParams P2 = mk_sa(1, 128, 128, 128, 128);
  SaParams P3 = mk_sa(2, 128, 128, 128, 128);

  // backbones
  run_sa(st, tmpl, nullptr, 512, 256, 0.3f, true,  P1, t_x1, t_f1, idx_buf, nidx_buf, mean, rstd, psum, psq, slabA, s16A, s16B);
  run_sa(st, t_x1, t_f1,    256, 128, 0.5f, false, P2, t_x2, t_f2, idx_buf, nidx_buf, mean, rstd, psum, psq, slabA, s16A, s16B);
  run_sa(st, t_x2, t_f2,    128, 64,  0.7f, false, P3, t_x3, t_f3, idx_buf, nidx_buf, mean, rstd, psum, psq, slabA, s16A, s16B);
  run_sa(st, srch, nullptr, 1024, 512, 0.3f, true,  P1, s_x1, s_f1, idx_buf, nidx_buf, mean, rstd, psum, psq, slabA, s16A, s16B);
  run_sa(st, s_x1, s_f1,    512,  256, 0.5f, false, P2, s_x2, s_f2, idx_buf, nidx_buf, mean, rstd, psum, psq, slabA, s16A, s16B);
  run_sa(st, s_x2, s_f2,    256,  128, 0.7f, false, P3, s_x3, s_f3, idx_buf, nidx_buf, mean, rstd, psum, psq, slabA, s16A, s16B);

  const int P = 128;
  // classification
  rows_to_stack_kernel<<<cdiv((long long)NN * 128, 256), 256, 0, st>>>(s_f3, NN, 128, stack16_a);
  {
    const float* lw[4] = {cla_lw[0], cla_lw[1], cla_lw[2], cla_lw[3]};
    const float* lb[4] = {cla_lb[0], cla_lb[1], cla_lb[2], cla_lb[3]};
    int lo[4] = {128, 128, 128, 1};
    run_stack(st, stack16_a, 128, NN, P, lw, lb, lo, 4, cla_bg, cla_bb,
              stack16_b, stack16_c, stackF, cla_y, mean, rstd);
  }
  sigmoid_kernel<<<cdiv(NN, 256), 256, 0, st>>>(cla_y, score, NN);

  // vote
  assemble_fx_kernel<<<cdiv((long long)NN * 131, 256), 256, 0, st>>>(s_x3, s_f3, P, stack16_a);
  {
    const float* lw[5] = {vote_lw[0], vote_lw[1], vote_lw[2], vote_lw[3], vote_lw[4]};
    const float* lb[5] = {vote_lb[0], vote_lb[1], vote_lb[2], vote_lb[3], vote_lb[4]};
    int lo[5] = {128, 128, 128, 128, 131};
    run_stack(st, stack16_a, 131, NN, P, lw, lb, lo, 5, vote_bg, vote_bb,
              stack16_b, stack16_c, stackF, off_y, mean, rstd);
  }
  temp_sel_kernel<<<cdiv(NN * 3, 256), 256, 0, st>>>(off_y, s_x3, P, ts_buf);
  fusion_update_kernel<<<cdiv((long long)128 * NN, 256), 256, 0, st>>>(off_y, s_f3, P, fusion);

  // pooled template / search descriptors
  ball_query_kernel<<<cdiv(BATCH * P, 128), 128, 0, st>>>(t_x3, ts_buf, 64, P, 8, 1.f, nidx_buf);
  group_max_kernel<<<cdiv(131 * BATCH * P, 256), 256, 0, st>>>(t_x3, ts_buf, t_f3, nidx_buf, 64, P, 8, tp_buf);
  ball_query_kernel<<<cdiv(BATCH * P, 128), 128, 0, st>>>(s_x3, s_x3, 128, P, 8, 1.f, nidx_buf);
  group_max_kernel<<<cdiv(131 * BATCH * P, 256), 256, 0, st>>>(s_x3, s_x3, s_f3, nidx_buf, 128, P, 8, sp_buf);

  // proposal
  assemble_prop_kernel<<<cdiv((long long)NN * 391, 256), 256, 0, st>>>(score, tp_buf, sp_buf, fusion, P, stack16_a);
  {
    const float* lw[6] = {prop_lw[0], prop_lw[1], prop_lw[2], prop_lw[3], prop_lw[4], prop_lw[5]};
    const float* lb[6] = {prop_lb[0], prop_lb[1], prop_lb[2], prop_lb[3], prop_lb[4], prop_lb[5]};
    int lo[6] = {128, 128, 128, 128, 128, 128};
    run_stack(st, stack16_a, 391, NN, P, lw, lb, lo, 6, prop_bg, prop_bb,
              stack16_b, stack16_c, stackF, prop_y, mean, rstd);
  }

  // voxelize
  long long voxn = (long long)131 * BATCH * 16416;
  zero_kernel<<<cdiv(voxn, 256), 256, 0, st>>>(slabA, voxn);
  zero_kernel<<<cdiv((long long)BATCH * 16416, 256), 256, 0, st>>>(cnt_buf, (long long)BATCH * 16416);
  vox_scatter_kernel<<<cdiv(BATCH * P * 132, 256), 256, 0, st>>>(prop_y, s_x3, P, slabA, cnt_buf);
  vox_norm_kernel<<<cdiv(voxn, 256), 256, 0, st>>>(slabA, cnt_buf);

  // CML 3-D convs
  {
    GemmP p{}; p.A = cml_cw[0]; p.Bsrc = slabA; p.bias = cml_cb[0]; p.C = slabB;
    p.M = 64; p.N = BATCH * 9 * 24 * 38; p.K = 131 * 27; p.mode = 2; p.outT = 0;
    p.ldC = p.N; p.Bn = BATCH; p.Din = 18; p.Hin = 24; p.Win = 38;
    p.Dout = 9; p.Hout = 24; p.Wout = 38; p.sd = 2; p.pd = 1;
    launch_gemm(st, p);
    bn_rows(st, slabB, 64, p.N, cml_bn_g[0], cml_bn_b[0], mean, rstd);
  }
  {
    GemmP p{}; p.A = cml_cw[1]; p.Bsrc = slabB; p.bias = cml_cb[1]; p.C = slabA;
    p.M = 64; p.N = BATCH * 7 * 24 * 38; p.K = 64 * 27; p.mode = 2; p.outT = 0;
    p.ldC = p.N; p.Bn = BATCH; p.Din = 9; p.Hin = 24; p.Win = 38;
    p.Dout = 7; p.Hout = 24; p.Wout = 38; p.sd = 1; p.pd = 0;
    launch_gemm(st, p);
    bn_rows(st, slabA, 64, p.N, cml_bn_g[1], cml_bn_b[1], mean, rstd);
  }
  {
    GemmP p{}; p.A = cml_cw[2]; p.Bsrc = slabA; p.bias = cml_cb[2]; p.C = slabB;
    p.M = 64; p.N = BATCH * 4 * 24 * 38; p.K = 64 * 27; p.mode = 2; p.outT = 0;
    p.ldC = p.N; p.Bn = BATCH; p.Din = 7; p.Hin = 24; p.Win = 38;
    p.Dout = 4; p.Hout = 24; p.Wout = 38; p.sd = 2; p.pd = 1;
    launch_gemm(st, p);
    bn_rows(st, slabB, 64, p.N, cml_bn_g[2], cml_bn_b[2], mean, rstd);
  }

  // RPN shared 2-D conv
  {
    GemmP p{}; p.A = sh_cw; p.Bsrc = slabB; p.bias = sh_cb; p.C = slabA;
    p.M = 128; p.N = BATCH * 24 * 38; p.K = 256 * 9; p.mode = 3; p.outT = 0;
    p.ldC = p.N; p.Bn = BATCH; p.Hin = 24; p.Win = 38;
    p.Hout = 24; p.Wout = 38; p.Dsplit = 4;
    launch_gemm(st, p);
    bn_rows(st, slabA, 128, p.N, sh_bg, sh_bb, mean, rstd);
  }

  rpn_heads_kernel<<<cdiv(BATCH * 5 * 912, 128), 128, 0, st>>>(
      slabA, hm_w, hm_b, loc_w, loc_b, z_w, z_b, (float*)d_out);
}
